// Transformer_14285061226647
// MI455X (gfx1250) — matrix operand order repaired and checked by reading, not compiled, not run on hardware
//
#include <hip/hip_runtime.h>
#include <hip/hip_bf16.h>

// ---------------------------------------------------------------------------
// MI455X (gfx1250, wave32) transformer-encoder forward.
// GEMMs + attention matmuls use v_wmma_f32_16x16x32_bf16; GEMM weight tiles
// are staged with GLOBAL_LOAD_ASYNC_TO_LDS_B128 (ASYNCcnt) double-buffering.
// ---------------------------------------------------------------------------

typedef __bf16 bf16_t;
typedef __attribute__((ext_vector_type(8)))  bf16_t v8bf;
typedef __attribute__((ext_vector_type(16))) bf16_t v16bf;
typedef __attribute__((ext_vector_type(8)))  float  v8f;

static constexpr int S_  = 2048;
static constexpr int D_  = 1024;
static constexpr int H_  = 16;
static constexpr int HD_ = 64;
static constexpr int F_  = 4096;
static constexpr int L_  = 4;
static constexpr int TD_ = 3072;   // 3*D
static constexpr int WIN_ = 128;

__device__ __forceinline__ v16bf cmb(v8bf lo, v8bf hi) {
  union { v16bf v; v8bf h[2]; } u;
  u.h[0] = lo; u.h[1] = hi;
  return u.v;
}

__device__ __forceinline__ v8f wmma_bf(v16bf a, v16bf b, v8f c) {
  // D = A(16x32 bf16) * B(32x16 bf16) + C(16x16 f32)
  return __builtin_amdgcn_wmma_f32_16x16x32_bf16(false, a, false, b,
                                                 (short)0, c, false, false);
}

// Async global -> LDS copy of 16 bytes per lane (tracked by ASYNCcnt).
__device__ __forceinline__ void async_b128(unsigned lds_off, const void* gptr) {
  asm volatile("global_load_async_to_lds_b128 %0, %1, off"
               :: "v"(lds_off), "v"(gptr) : "memory");
}
__device__ __forceinline__ void wait_async0() {
  asm volatile("s_wait_asynccnt 0x0" ::: "memory");
}

// ---------------------------------------------------------------------------
// x = X + sinusoidal positional encoding
// ---------------------------------------------------------------------------
__global__ __launch_bounds__(256) void posenc_kernel(const float* __restrict__ X,
                                                     float* __restrict__ x) {
  int i = blockIdx.x * 256 + threadIdx.x;        // over S*D
  int s = i >> 10;
  int d = i & (D_ - 1);
  int pair = d >> 1;
  float div = __expf((float)(2 * pair) * (-9.210340371976184f / (float)D_));
  float ang = (float)s * div;
  float pe = (d & 1) ? __cosf(ang) : __sinf(ang);
  x[i] = X[i] + pe;
}

// ---------------------------------------------------------------------------
// Tiled transpose + fp32->bf16 convert:  dst[c][r] = (bf16) src[r*ld + c]
// grid = (C/32, R/32), block = 256
// ---------------------------------------------------------------------------
__global__ __launch_bounds__(256) void tcvt_kernel(const float* __restrict__ src,
                                                   bf16_t* __restrict__ dst,
                                                   int R, int C, int ld) {
  __shared__ float tile[32][33];
  int bc = blockIdx.x * 32;     // col of src (dst-major dim)
  int br = blockIdx.y * 32;     // row of src
  int tx = threadIdx.x & 31;
  int ty = threadIdx.x >> 5;    // 0..7
  for (int i = 0; i < 32; i += 8)
    tile[ty + i][tx] = src[(size_t)(br + ty + i) * ld + bc + tx];
  __syncthreads();
  for (int i = 0; i < 32; i += 8)
    dst[(size_t)(bc + ty + i) * R + br + tx] = (bf16_t)tile[tx][ty + i];
}

// ---------------------------------------------------------------------------
// Elementwise fp32 -> bf16
// ---------------------------------------------------------------------------
__global__ __launch_bounds__(256) void cvt_kernel(const float* __restrict__ in,
                                                  bf16_t* __restrict__ out) {
  int i = blockIdx.x * 256 + threadIdx.x;
  out[i] = (bf16_t)in[i];
}

// ---------------------------------------------------------------------------
// GEMM: C[M][N] = A[M][K](fp32, converted in-kernel) * Bt[K][N](bf16) + bias
// Block: 256 threads = 8 waves; tile 128x128x32; wave does 64x32 (4x2 WMMA).
// Double-buffered LDS; B staged with async-to-LDS, A via regs + cvt.
// ---------------------------------------------------------------------------
__global__ __launch_bounds__(256) void gemm_bf16_kernel(
    const float* __restrict__ A, const bf16_t* __restrict__ Bt,
    const float* __restrict__ bias, float* __restrict__ C,
    int M, int N, int K, int do_relu) {
  constexpr int BM = 128, BN = 128, BK = 32;
  __shared__ bf16_t sA[2][BM][BK];   // 16 KB
  __shared__ bf16_t sB[2][BK][BN];   // 16 KB

  const int tid  = threadIdx.x;
  const int lane = tid & 31;
  const int wave = tid >> 5;         // 0..7
  const int wm   = wave & 1;         // row half (64 rows)
  const int wn   = wave >> 1;        // col quarter (32 cols)
  const int m0   = blockIdx.y * BM;
  const int n0   = blockIdx.x * BN;

  const int ml = lane & 15;          // A frag row / C col
  const int kg = lane >> 4;          // A frag k-group / C row group

  // staging assignment
  const int ar = tid >> 1;           // 0..127  (A row)
  const int ac = (tid & 1) * 16;     // 0 or 16 (A col)
  const int br = tid >> 3;           // 0..31   (B row)
  const int bc = (tid & 7) * 16;     // 0..112  (B col)

  const float*  agp = A  + (size_t)(m0 + ar) * K + ac;   // + ks*BK
  const bf16_t* bgp = Bt + (size_t)br * N + n0 + bc;     // + ks*BK*N

  v8f acc[4][2] = {};

  auto stageB = [&](int ks, int buf) {
    unsigned lo = (unsigned)(uintptr_t)&sB[buf][br][bc];
    const bf16_t* g = bgp + (size_t)ks * BK * N;
    async_b128(lo, g);
    async_b128(lo + 16u, g + 8);
  };
  auto loadA = [&](int ks, float4* r) {
    const float4* ap = (const float4*)(agp + (size_t)ks * BK);
    r[0] = ap[0]; r[1] = ap[1]; r[2] = ap[2]; r[3] = ap[3];
  };
  auto storeA = [&](int buf, const float4* r) {
    union { v8bf v; bf16_t e[8]; } u0, u1;
    u0.e[0] = (bf16_t)r[0].x; u0.e[1] = (bf16_t)r[0].y;
    u0.e[2] = (bf16_t)r[0].z; u0.e[3] = (bf16_t)r[0].w;
    u0.e[4] = (bf16_t)r[1].x; u0.e[5] = (bf16_t)r[1].y;
    u0.e[6] = (bf16_t)r[1].z; u0.e[7] = (bf16_t)r[1].w;
    u1.e[0] = (bf16_t)r[2].x; u1.e[1] = (bf16_t)r[2].y;
    u1.e[2] = (bf16_t)r[2].z; u1.e[3] = (bf16_t)r[2].w;
    u1.e[4] = (bf16_t)r[3].x; u1.e[5] = (bf16_t)r[3].y;
    u1.e[6] = (bf16_t)r[3].z; u1.e[7] = (bf16_t)r[3].w;
    *(v8bf*)&sA[buf][ar][ac]     = u0.v;
    *(v8bf*)&sA[buf][ar][ac + 8] = u1.v;
  };

  // ---- prologue: stage slab 0 into buffer 0 ----
  {
    float4 r[4];
    stageB(0, 0);
    loadA(0, r);
    storeA(0, r);
  }
  wait_async0();
  __syncthreads();

  const int nk = K / BK;
  for (int ks = 0; ks < nk; ++ks) {
    const int cur = ks & 1;
    const int nxt = cur ^ 1;
    const bool more = (ks + 1 < nk);

    float4 r[4];
    if (more) {
      stageB(ks + 1, nxt);         // async B -> LDS[next], overlaps compute
      loadA(ks + 1, r);            // A slab -> regs, overlaps compute
      if (ks + 2 < nk)
        __builtin_prefetch(agp + (size_t)(ks + 2) * BK, 0, 1);
    }

    // ---- compute on current buffer ----
    v16bf bfrag[2];
    for (int j = 0; j < 2; ++j) {
      const bf16_t* pb = &sB[cur][kg * 8][wn * 32 + j * 16 + ml];   // lane <-> N col
      for (int s = 0; s < 16; ++s) bfrag[j][s] = pb[(s < 8 ? s : s + 8) * BN];
    }
    for (int i = 0; i < 4; ++i) {
      const bf16_t* pa = &sA[cur][wm * 64 + i * 16 + ml][kg * 8];
      v16bf afrag = cmb(*(const v8bf*)pa, *(const v8bf*)(pa + 16));
      for (int j = 0; j < 2; ++j)
        acc[i][j] = wmma_bf(afrag, bfrag[j], acc[i][j]);
    }

    if (more) storeA(nxt, r);      // cvt fp32->bf16, ds_store into LDS[next]
    wait_async0();                 // our async stores to LDS[next] complete
    __syncthreads();               // publish next buffer / retire current
  }

  // ---- store: C layout: VGPR r -> M = r + 8*kg, N = ml ----
  for (int i = 0; i < 4; ++i) {
    int mbase = m0 + wm * 64 + i * 16 + kg * 8;
    for (int j = 0; j < 2; ++j) {
      int ncol = n0 + wn * 32 + j * 16 + ml;
      float bv = bias ? bias[ncol] : 0.0f;
      float* cp = C + (size_t)mbase * N + ncol;
      for (int r = 0; r < 8; ++r) {
        float v = acc[i][j][r] + bv;
        if (do_relu) v = fmaxf(v, 0.0f);
        cp[(size_t)r * N] = v;
      }
    }
  }
}

// ---------------------------------------------------------------------------
// Banded attention. One block (128 thr = 4 waves) per (query-tile of 16, head).
// Strip = 17 t-tiles (272 cols), padded to 288 for 9 K=32 steps in P*V.
// ---------------------------------------------------------------------------
__global__ __launch_bounds__(128) void attn_kernel(
    const bf16_t* __restrict__ qkv_bf,   // [S][3072] bf16
    const bf16_t* __restrict__ kT,       // [H*64][S]  bf16
    float* __restrict__ out) {           // [S][1024]  fp32
  constexpr int NT = 17;
  constexpr int STRIP = NT * 16;     // 272
  constexpr int PCOLS = 288;         // 9 * 32
  __shared__ float  sS[16][STRIP];   // 17 KB
  __shared__ bf16_t sP[16][PCOLS];   // 9 KB

  const int h    = blockIdx.y;
  const int qs0  = blockIdx.x * 16;
  const int t0   = qs0 - WIN_;          // strip origin (multiple of 16, may be <0)
  const int tid  = threadIdx.x;
  const int lane = tid & 31;
  const int wave = tid >> 5;            // 0..3
  const int ml   = lane & 15;
  const int kg   = lane >> 4;

  // --- q fragments (A layout), d-chunks 0 and 32 ---
  v16bf qf[2];
  {
    const bf16_t* qrow = qkv_bf + (size_t)(qs0 + ml) * TD_ + h * HD_;
    for (int kd = 0; kd < 2; ++kd) {
      const bf16_t* p = qrow + kd * 32 + kg * 8;
      qf[kd] = cmb(*(const v8bf*)p, *(const v8bf*)(p + 16));
    }
  }

  // --- Pass A: scores = q . k^T, scale + band mask -> sS ---
  for (int nt = wave; nt < NT; nt += 4) {
    int tb = t0 + nt * 16;
    int tcl = tb < 0 ? 0 : (tb > S_ - 16 ? S_ - 16 : tb);   // clamp (masked later)
    v8f acc = {};
    for (int kd = 0; kd < 2; ++kd) {
      const bf16_t* kp = qkv_bf + (size_t)(tcl + ml) * TD_ + D_ + h * HD_ + kd * 32 + kg * 8;
      v16bf bfg = cmb(*(const v8bf*)kp, *(const v8bf*)(kp + 16));
      acc = wmma_bf(qf[kd], bfg, acc);
    }
    for (int r = 0; r < 8; ++r) {
      int m = r + 8 * kg;
      int trel = nt * 16 + ml;
      int t = tb + ml;
      int s_abs = qs0 + m;
      int di = s_abs - t;
      bool ok = (t >= 0) && (t < S_) && (di < WIN_) && (di > -WIN_);
      sS[m][trel] = ok ? acc[r] * 0.125f : -1.0e30f;
    }
  }
  __syncthreads();

  // --- Pass B: row softmax, write normalized probs (bf16, zero-padded) ---
  for (int i = 0; i < 4; ++i) {
    int row = wave * 4 + i;
    float mx = -1.0e30f;
    for (int c = lane; c < STRIP; c += 32) mx = fmaxf(mx, sS[row][c]);
    for (int off = 16; off; off >>= 1) mx = fmaxf(mx, __shfl_xor(mx, off, 32));
    float sum = 0.0f;
    for (int c = lane; c < STRIP; c += 32) {
      float e = __expf(sS[row][c] - mx);
      sS[row][c] = e;
      sum += e;
    }
    for (int off = 16; off; off >>= 1) sum += __shfl_xor(sum, off, 32);
    float inv = 1.0f / sum;
    for (int c = lane; c < PCOLS; c += 32)
      sP[row][c] = (c < STRIP) ? (bf16_t)(sS[row][c] * inv) : (bf16_t)0.0f;
  }
  __syncthreads();

  // --- Pass C: O = P . V ; wave owns d-tile `wave` (16 cols of hd=64) ---
  v8f oacc = {};
  for (int kt = 0; kt < 9; ++kt) {
    const bf16_t* pp = &sP[ml][kt * 32 + kg * 8];
    v16bf af = cmb(*(const v8bf*)pp, *(const v8bf*)(pp + 16));
    v16bf bfg;
    for (int s = 0; s < 16; ++s) {
      int t = t0 + kt * 32 + kg * 8 + (s < 8 ? s : s + 8);
      t = t < 0 ? 0 : (t > S_ - 1 ? S_ - 1 : t);       // P==0 on OOB cols
      bfg[s] = qkv_bf[(size_t)t * TD_ + 2 * D_ + h * HD_ + wave * 16 + ml];
    }
    oacc = wmma_bf(af, bfg, oacc);
  }
  for (int r = 0; r < 8; ++r) {
    int s = qs0 + r + 8 * kg;
    out[(size_t)s * D_ + h * HD_ + wave * 16 + ml] = oacc[r];
  }
}

// ---------------------------------------------------------------------------
// x = LayerNorm(x + addend) * g + b        (block per row, 256 threads)
// ---------------------------------------------------------------------------
__global__ __launch_bounds__(256) void ln_res_kernel(
    float* __restrict__ x, const float* __restrict__ addend,
    const float* __restrict__ g, const float* __restrict__ b) {
  const int row = blockIdx.x;
  const int tid = threadIdx.x;
  float v[4];
  float sum = 0.0f, sq = 0.0f;
  const float* xr = x + (size_t)row * D_;
  const float* ar = addend + (size_t)row * D_;
  for (int i = 0; i < 4; ++i) {
    float t = xr[tid + 256 * i] + ar[tid + 256 * i];
    v[i] = t; sum += t; sq += t * t;
  }
  for (int off = 16; off; off >>= 1) {
    sum += __shfl_xor(sum, off, 32);
    sq  += __shfl_xor(sq,  off, 32);
  }
  __shared__ float ws0[8], ws1[8];
  if ((tid & 31) == 0) { ws0[tid >> 5] = sum; ws1[tid >> 5] = sq; }
  __syncthreads();
  float s = 0.0f, q = 0.0f;
  for (int w = 0; w < 8; ++w) { s += ws0[w]; q += ws1[w]; }
  float mu  = s * (1.0f / D_);
  float var = q * (1.0f / D_) - mu * mu;
  float inv = rsqrtf(var + 1e-5f);
  float* xw = x + (size_t)row * D_;
  for (int i = 0; i < 4; ++i) {
    int c = tid + 256 * i;
    xw[c] = (v[i] - mu) * inv * g[c] + b[c];
  }
}

// ---------------------------------------------------------------------------
// logits + log_softmax over 3 classes. One wave per row.
// ---------------------------------------------------------------------------
__global__ __launch_bounds__(256) void decode_kernel(
    const float* __restrict__ x, const float* __restrict__ Wdec,
    const float* __restrict__ bdec, float* __restrict__ out) {
  const int row  = blockIdx.x * 8 + (threadIdx.x >> 5);
  const int lane = threadIdx.x & 31;
  const float* xr = x + (size_t)row * D_;
  float a0 = 0.0f, a1 = 0.0f, a2 = 0.0f;
  for (int c = lane; c < D_; c += 32) {
    float xv = xr[c];
    a0 += xv * Wdec[c];
    a1 += xv * Wdec[D_ + c];
    a2 += xv * Wdec[2 * D_ + c];
  }
  for (int off = 16; off; off >>= 1) {
    a0 += __shfl_xor(a0, off, 32);
    a1 += __shfl_xor(a1, off, 32);
    a2 += __shfl_xor(a2, off, 32);
  }
  if (lane == 0) {
    float l0 = a0 + bdec[0], l1 = a1 + bdec[1], l2 = a2 + bdec[2];
    float mx = fmaxf(l0, fmaxf(l1, l2));
    float lse = mx + __logf(__expf(l0 - mx) + __expf(l1 - mx) + __expf(l2 - mx));
    out[(size_t)row * 3 + 0] = l0 - lse;
    out[(size_t)row * 3 + 1] = l1 - lse;
    out[(size_t)row * 3 + 2] = l2 - lse;
  }
}

// ---------------------------------------------------------------------------
// Host orchestration
// ---------------------------------------------------------------------------
extern "C" void kernel_launch(void* const* d_in, const int* in_sizes, int n_in,
                              void* d_out, int out_size, void* d_ws, size_t ws_size,
                              hipStream_t stream) {
  const float* X     = (const float*)d_in[0];
  const float* Wqkv  = (const float*)d_in[1];
  const float* bqkv  = (const float*)d_in[2];
  const float* Wo    = (const float*)d_in[3];
  const float* bo    = (const float*)d_in[4];
  const float* ln1_g = (const float*)d_in[5];
  const float* ln1_b = (const float*)d_in[6];
  const float* W1    = (const float*)d_in[7];
  const float* b1    = (const float*)d_in[8];
  const float* W2    = (const float*)d_in[9];
  const float* b2    = (const float*)d_in[10];
  const float* ln2_g = (const float*)d_in[11];
  const float* ln2_b = (const float*)d_in[12];
  const float* Wdec  = (const float*)d_in[13];
  const float* bdec  = (const float*)d_in[14];
  (void)in_sizes; (void)n_in; (void)out_size; (void)ws_size;

  // workspace layout (bytes)
  char* base = (char*)d_ws;
  size_t off = 0;
  auto carve = [&](size_t bytes) { void* p = base + off; off += (bytes + 255) & ~(size_t)255; return p; };
  float*  x       = (float*)carve((size_t)S_ * D_ * 4);       // 8 MB
  float*  qkv     = (float*)carve((size_t)S_ * TD_ * 4);      // 24 MB
  bf16_t* qkv_bf  = (bf16_t*)carve((size_t)S_ * TD_ * 2);     // 12 MB
  bf16_t* kT      = (bf16_t*)carve((size_t)D_ * S_ * 2);      // 4 MB
  float*  attnout = (float*)carve((size_t)S_ * D_ * 4);       // 8 MB
  float*  o       = (float*)carve((size_t)S_ * D_ * 4);       // 8 MB
  float*  hbuf    = (float*)carve((size_t)S_ * F_ * 4);       // 32 MB
  float*  ff      = (float*)carve((size_t)S_ * D_ * 4);       // 8 MB
  bf16_t* wT      = (bf16_t*)carve((size_t)F_ * D_ * 2);      // 8 MB (reused)

  dim3 blk256(256), blk128(128);

  posenc_kernel<<<(S_ * D_) / 256, blk256, 0, stream>>>(X, x);

  for (int l = 0; l < L_; ++l) {
    // --- QKV projection ---
    tcvt_kernel<<<dim3(D_ / 32, TD_ / 32), blk256, 0, stream>>>(
        Wqkv + (size_t)l * TD_ * D_, wT, TD_, D_, D_);
    gemm_bf16_kernel<<<dim3(TD_ / 128, S_ / 128), blk256, 0, stream>>>(
        x, wT, bqkv + (size_t)l * TD_, qkv, S_, TD_, D_, 0);
    cvt_kernel<<<(S_ * TD_) / 256, blk256, 0, stream>>>(qkv, qkv_bf);
    // k^T: src rows = t (ld 3072, offset D_), cols = h*64+d  ->  kT[d_lin][t]
    tcvt_kernel<<<dim3(D_ / 32, S_ / 32), blk256, 0, stream>>>(
        qkv + D_, kT, S_, D_, TD_);

    // --- banded attention ---
    attn_kernel<<<dim3(S_ / 16, H_), blk128, 0, stream>>>(qkv_bf, kT, attnout);

    // --- output projection + residual LN ---
    tcvt_kernel<<<dim3(D_ / 32, D_ / 32), blk256, 0, stream>>>(
        Wo + (size_t)l * D_ * D_, wT, D_, D_, D_);
    gemm_bf16_kernel<<<dim3(D_ / 128, S_ / 128), blk256, 0, stream>>>(
        attnout, wT, bo + (size_t)l * D_, o, S_, D_, D_, 0);
    ln_res_kernel<<<S_, blk256, 0, stream>>>(x, o, ln1_g + (size_t)l * D_,
                                             ln1_b + (size_t)l * D_);

    // --- feed-forward ---
    tcvt_kernel<<<dim3(D_ / 32, F_ / 32), blk256, 0, stream>>>(
        W1 + (size_t)l * F_ * D_, wT, F_, D_, D_);
    gemm_bf16_kernel<<<dim3(F_ / 128, S_ / 128), blk256, 0, stream>>>(
        x, wT, b1 + (size_t)l * F_, hbuf, S_, F_, D_, 1);
    tcvt_kernel<<<dim3(F_ / 32, D_ / 32), blk256, 0, stream>>>(
        W2 + (size_t)l * D_ * F_, wT, D_, F_, F_);
    gemm_bf16_kernel<<<dim3(D_ / 128, S_ / 128), blk256, 0, stream>>>(
        hbuf, wT, b2 + (size_t)l * D_, ff, S_, D_, F_, 0);
    ln_res_kernel<<<S_, blk256, 0, stream>>>(x, ff, ln2_g + (size_t)l * D_,
                                             ln2_b + (size_t)l * D_);
  }

  decode_kernel<<<S_ / 8, blk256, 0, stream>>>(x, Wdec, bdec, (float*)d_out);
}
